// GraphSAGE_GraphSVX_foroptuna_51402168598679
// MI455X (gfx1250) — compile-verified, matrix operand order checked by reading, not run on hardware
//
#include <hip/hip_runtime.h>
#include <stdint.h>

#define HF 128          // hidden / feature width (F_IN == H == 128)
#define EPSV 1e-5f

typedef __attribute__((ext_vector_type(16))) __bf16 v16bf;
typedef __attribute__((ext_vector_type(8)))  float  v8f;

// ---------- helpers ----------
__device__ __forceinline__ unsigned short f2bf(float f) {
  unsigned u = __float_as_uint(f);
  unsigned r = 0x7FFFu + ((u >> 16) & 1u);       // round-to-nearest-even
  return (unsigned short)((u + r) >> 16);
}
__device__ __forceinline__ uint2 pack4(float4 v) {
  uint2 o;
  o.x = ((unsigned)f2bf(v.y) << 16) | (unsigned)f2bf(v.x);
  o.y = ((unsigned)f2bf(v.w) << 16) | (unsigned)f2bf(v.z);
  return o;
}

// ---------- f32 -> bf16 conversion (vectorized: 4 elems / thread) ----------
__global__ void cvt_f32_bf16(const float* __restrict__ in,
                             unsigned short* __restrict__ out, int n4) {
  int gid = blockIdx.x * blockDim.x + threadIdx.x;
  if (gid >= n4) return;
  float4 v = ((const float4*)in)[gid];
  ((uint2*)out)[gid] = pack4(v);
}

// ---------- zero fill ----------
__global__ void zero_f32(float* __restrict__ p, int n) {
  int gid = blockIdx.x * blockDim.x + threadIdx.x;
  if (gid < n) p[gid] = 0.0f;
}

// ---------- edge scatter: one wave per edge, float4 per lane ----------
__global__ void scatter_kernel(const float* __restrict__ h,
                               const int* __restrict__ src,
                               const int* __restrict__ dst,
                               float* __restrict__ msg,
                               float* __restrict__ deg, int E) {
  int t = blockIdx.x * blockDim.x + threadIdx.x;
  int wid = t >> 5, lane = t & 31;
  if (wid >= E) return;
  int s = src[wid], d = dst[wid];
  float4 v = ((const float4*)(h + (size_t)s * HF))[lane];
  float* mb = msg + (size_t)d * HF + lane * 4;
  atomicAdd(mb + 0, v.x);
  atomicAdd(mb + 1, v.y);
  atomicAdd(mb + 2, v.z);
  atomicAdd(mb + 3, v.w);
  if (lane == 0) atomicAdd(deg + d, 1.0f);
}

// ---------- agg = msg/max(deg,1) -> bf16, and h -> bf16 (fused) ----------
__global__ void agg_div_cvt(const float* __restrict__ msg,
                            const float* __restrict__ deg,
                            const float* __restrict__ h,
                            unsigned short* __restrict__ aggbf,
                            unsigned short* __restrict__ hbf, int Nn) {
  int gid = blockIdx.x * blockDim.x + threadIdx.x;
  if (gid >= Nn * 32) return;
  int n = gid >> 5;
  float inv = 1.0f / fmaxf(deg[n], 1.0f);
  float4 m = ((const float4*)msg)[gid];
  m.x *= inv; m.y *= inv; m.z *= inv; m.w *= inv;
  ((uint2*)aggbf)[gid] = pack4(m);
  ((uint2*)hbf)[gid]   = pack4(((const float4*)h)[gid]);
}

// ---------- WMMA fragment loader ----------
// bf16 16x32 A/B fragment: lane = {row/col = lane&15, half = lane>>4};
// dword p (p=0..3) holds K = k0 + 2p + 8*half, dword 4+p holds K = k0+16+2p+8*half.
__device__ __forceinline__ v16bf load_frag(const unsigned* __restrict__ row,
                                           int k0, int half) {
  union { v16bf v; unsigned u[8]; } f;
  int b0 = (k0 >> 1) + 4 * half;
#pragma unroll
  for (int p = 0; p < 4; ++p) {
    f.u[p]     = row[b0 + p];
    f.u[4 + p] = row[b0 + 8 + p];
  }
  return f.v;
}

// ---------- dual WMMA GEMM: out = A@Wl^T (+ G@Wr^T) + bias ----------
// A,G: [N][128] bf16.  Wl,Wr: [128][128] bf16 row-major (out-feature major ==
// B^T, so no transpose needed).  Block = 256 thr = 8 waves; wave w owns cols
// [16w,16w+16) of a 16-row strip. A/G tiles staged in LDS (pitch 65 dwords to
// avoid 64-bank conflicts); weights read from global (WGP$-resident, 32KB).
__global__ void __launch_bounds__(256)
wmma_gemm_dual(const unsigned short* __restrict__ Abf,
               const unsigned short* __restrict__ Gbf,
               const unsigned short* __restrict__ Wl,
               const unsigned short* __restrict__ Wr,
               const float* __restrict__ bias,
               float* __restrict__ out, int Nrows) {
  __shared__ unsigned sA[16 * 65];
  __shared__ unsigned sG[16 * 65];
  const int tid = threadIdx.x;
  const int m0  = blockIdx.x * 16;
  const bool dual = (Gbf != nullptr);
  const size_t maxdw = (size_t)Nrows * 64;
  {
    const unsigned* gA = (const unsigned*)Abf + (size_t)m0 * 64;
    for (int i = tid; i < 1024; i += 256) {
      int r = i >> 6, c = i & 63;
      sA[r * 65 + c] = ((size_t)m0 * 64 + i < maxdw) ? gA[i] : 0u;
    }
    if (dual) {
      const unsigned* gG = (const unsigned*)Gbf + (size_t)m0 * 64;
      for (int i = tid; i < 1024; i += 256) {
        int r = i >> 6, c = i & 63;
        sG[r * 65 + c] = ((size_t)m0 * 64 + i < maxdw) ? gG[i] : 0u;
      }
    }
  }
  __syncthreads();

  const int wave = tid >> 5;
  const int lane = tid & 31;
  const int l16  = lane & 15;
  const int half = lane >> 4;
  const int n0   = wave * 16;

  v8f acc = {};
  const unsigned* arow  = sA + l16 * 65;
  const unsigned* wlrow = (const unsigned*)Wl + (size_t)(n0 + l16) * 64;
#pragma unroll
  for (int k0 = 0; k0 < 128; k0 += 32) {
    v16bf a = load_frag(arow, k0, half);
    v16bf b = load_frag(wlrow, k0, half);
    acc = __builtin_amdgcn_wmma_f32_16x16x32_bf16(false, a, false, b,
                                                  (short)0, acc, false, false);
  }
  if (dual) {
    const unsigned* grow  = sG + l16 * 65;
    const unsigned* wrrow = (const unsigned*)Wr + (size_t)(n0 + l16) * 64;
#pragma unroll
    for (int k0 = 0; k0 < 128; k0 += 32) {
      v16bf a = load_frag(grow, k0, half);
      v16bf b = load_frag(wrrow, k0, half);
      acc = __builtin_amdgcn_wmma_f32_16x16x32_bf16(false, a, false, b,
                                                    (short)0, acc, false, false);
    }
  }
  // C/D layout: VGPR r -> M = r + 8*half, N = lane&15
  const int col   = n0 + l16;
  const float bv  = bias[col];
  const int rbase = m0 + half * 8;
#pragma unroll
  for (int r = 0; r < 8; ++r) {
    int row = rbase + r;
    if (row < Nrows) out[(size_t)row * HF + col] = acc[r] + bv;
  }
}

// ---------- BN stats: per-block column partials + one atomic per column ----------
__global__ void bn_stats(const float* __restrict__ x, float* __restrict__ sum,
                         float* __restrict__ sumsq, int Nn) {
  int f  = threadIdx.x;              // 128 threads = 128 columns
  int r0 = blockIdx.x * 256;
  int r1 = min(r0 + 256, Nn);
  float s = 0.0f, ss = 0.0f;
  for (int r = r0; r < r1; ++r) {
    float v = x[(size_t)r * HF + f];
    s += v; ss += v * v;
  }
  atomicAdd(&sum[f], s);
  atomicAdd(&sumsq[f], ss);
}

// ---------- BN normalize + ReLU ----------
__global__ void bn_apply(const float* __restrict__ x,
                         const float* __restrict__ sum,
                         const float* __restrict__ sumsq,
                         const float* __restrict__ gam,
                         const float* __restrict__ bet,
                         float* __restrict__ y, int Nn) {
  int gid = blockIdx.x * blockDim.x + threadIdx.x;
  if (gid >= Nn * 32) return;
  int c = gid & 31;
  float invN = 1.0f / (float)Nn;
  union { float4 v; float a[4]; } in, outv;
  in.v = ((const float4*)x)[gid];
#pragma unroll
  for (int j = 0; j < 4; ++j) {
    int f = c * 4 + j;
    float mu  = sum[f] * invN;
    float var = fmaf(-mu, mu, sumsq[f] * invN);   // biased variance
    float sc  = rsqrtf(var + EPSV) * gam[f];
    float sh  = fmaf(-mu, sc, bet[f]);
    outv.a[j] = fmaxf(fmaf(in.a[j], sc, sh), 0.0f);
  }
  ((float4*)y)[gid] = outv.v;
}

// ---------- classifier: one wave per node, shuffle reduction ----------
__global__ void cls_kernel(const float* __restrict__ h,
                           const float* __restrict__ w,
                           const float* __restrict__ b,
                           float* __restrict__ out, int Nn) {
  int t = blockIdx.x * blockDim.x + threadIdx.x;
  int wid = t >> 5, lane = t & 31;
  if (wid >= Nn) return;
  float4 hv = ((const float4*)(h + (size_t)wid * HF))[lane];
  float4 w0 = ((const float4*)w)[lane];
  float4 w1 = ((const float4*)(w + HF))[lane];
  float d0 = hv.x * w0.x + hv.y * w0.y + hv.z * w0.z + hv.w * w0.w;
  float d1 = hv.x * w1.x + hv.y * w1.y + hv.z * w1.z + hv.w * w1.w;
#pragma unroll
  for (int off = 16; off > 0; off >>= 1) {
    d0 += __shfl_down(d0, off, 32);
    d1 += __shfl_down(d1, off, 32);
  }
  if (lane == 0) {
    out[(size_t)wid * 2 + 0] = d0 + b[0];
    out[(size_t)wid * 2 + 1] = d1 + b[1];
  }
}

// ---------- launcher ----------
extern "C" void kernel_launch(void* const* d_in, const int* in_sizes, int n_in,
                              void* d_out, int out_size, void* d_ws, size_t ws_size,
                              hipStream_t stream) {
  const float* x     = (const float*)d_in[0];
  const int*   ei    = (const int*)  d_in[1];
  const float* emb_w = (const float*)d_in[2];
  const float* emb_b = (const float*)d_in[3];
  const float* wl[3] = {(const float*)d_in[4],  (const float*)d_in[7],  (const float*)d_in[10]};
  const float* wr[3] = {(const float*)d_in[5],  (const float*)d_in[8],  (const float*)d_in[11]};
  const float* bb[3] = {(const float*)d_in[6],  (const float*)d_in[9],  (const float*)d_in[12]};
  const float* g[3]  = {(const float*)d_in[13], (const float*)d_in[15], (const float*)d_in[17]};
  const float* be[3] = {(const float*)d_in[14], (const float*)d_in[16], (const float*)d_in[18]};
  const float* cw    = (const float*)d_in[19];
  const float* cb    = (const float*)d_in[20];

  const int N_ = in_sizes[0] / HF;
  const int E_ = in_sizes[1] / 2;
  const int* srcI = ei;
  const int* dstI = ei + E_;

  // workspace carve-out (~155 MB): GEMM output reuses msg after agg extracted
  char* ws = (char*)d_ws; size_t off = 0;
  auto alloc = [&](size_t bytes) -> void* {
    void* p = ws + off; off += (bytes + 255) & ~(size_t)255; return p;
  };
  float* hA  = (float*)alloc((size_t)N_ * HF * 4);   // activations
  float* msg = (float*)alloc((size_t)N_ * HF * 4);   // scatter accum / GEMM out
  unsigned short* hbf   = (unsigned short*)alloc((size_t)N_ * HF * 2);
  unsigned short* aggbf = (unsigned short*)alloc((size_t)N_ * HF * 2);
  float* deg = (float*)alloc((size_t)N_ * 4);
  unsigned short* wbf[7];
  for (int i = 0; i < 7; ++i) wbf[i] = (unsigned short*)alloc((size_t)HF * HF * 2);
  float* stats = (float*)alloc(256 * 4);             // sum[128] | sumsq[128]

  const int TB = 256;
  const int n4  = N_ * 32;                           // float4 chunks in [N,128]
  const int gN4 = (n4 + TB - 1) / TB;
  const int gGemm = (N_ + 15) / 16;

  // bf16 mirrors of x and weights
  cvt_f32_bf16<<<gN4, TB, 0, stream>>>(x, hbf, n4);
  const float* wsrc[7] = {emb_w, wl[0], wr[0], wl[1], wr[1], wl[2], wr[2]};
  for (int i = 0; i < 7; ++i)
    cvt_f32_bf16<<<(4096 + TB - 1) / TB, TB, 0, stream>>>(wsrc[i], wbf[i], 4096);

  // h = x @ emb_w^T + emb_b
  wmma_gemm_dual<<<gGemm, TB, 0, stream>>>(hbf, nullptr, wbf[0], nullptr,
                                           emb_b, hA, N_);

  for (int L = 0; L < 3; ++L) {
    zero_f32<<<(N_ * HF + TB - 1) / TB, TB, 0, stream>>>(msg, N_ * HF);
    zero_f32<<<(N_ + TB - 1) / TB, TB, 0, stream>>>(deg, N_);
    scatter_kernel<<<((E_ * 32) + TB - 1) / TB, TB, 0, stream>>>(
        hA, srcI, dstI, msg, deg, E_);
    agg_div_cvt<<<gN4, TB, 0, stream>>>(msg, deg, hA, aggbf, hbf, N_);
    // out = h@wl^T + agg@wr^T + b   (into msg, which is now dead)
    wmma_gemm_dual<<<gGemm, TB, 0, stream>>>(hbf, aggbf, wbf[1 + 2 * L],
                                             wbf[2 + 2 * L], bb[L], msg, N_);
    zero_f32<<<1, 256, 0, stream>>>(stats, 256);
    bn_stats<<<(N_ + 255) / 256, 128, 0, stream>>>(msg, stats, stats + 128, N_);
    bn_apply<<<gN4, TB, 0, stream>>>(msg, stats, stats + 128, g[L], be[L], hA, N_);
  }

  cls_kernel<<<((N_ * 32) + TB - 1) / TB, TB, 0, stream>>>(hA, cw, cb,
                                                           (float*)d_out, N_);
}
